// AdaTTWSharedExps_7705171329620
// MI455X (gfx1250) — compile-verified, matrix operand order checked by reading, not run on hardware
//
#include <hip/hip_runtime.h>
#include <hip/hip_bf16.h>

// ---------------------------------------------------------------------------
// AdaTT with shared experts: 2 layers of (18-expert FFN + gate + combine).
// B=1024, D=512, H=1024, T=8, E=18, M0=9, M1=8.  EXPERT_IDX[e] = e>>1.
// Heavy GEMMs: v_wmma_f32_16x16x32_bf16.  FFN2 A-tile staged via TDM
// (tensor_load_to_lds) since H1 is already bf16 -> pure DMA, LDS pad via D#.
// ---------------------------------------------------------------------------

typedef __bf16 v16bf __attribute__((ext_vector_type(16)));
typedef float  v8f   __attribute__((ext_vector_type(8)));
typedef unsigned int u32x4 __attribute__((ext_vector_type(4)));
typedef int          i32x8 __attribute__((ext_vector_type(8)));
typedef int          i32x4 __attribute__((ext_vector_type(4)));

#if defined(__HIP_DEVICE_COMPILE__) && __has_builtin(__builtin_amdgcn_tensor_load_to_lds)
#define HAVE_TDM 1
#else
#define HAVE_TDM 0
#endif

__device__ __forceinline__ unsigned short f2bf(float f) {
    unsigned u = __float_as_uint(f);
    u += 0x7FFFu + ((u >> 16) & 1u);           // round-to-nearest-even
    return (unsigned short)(u >> 16);
}

static constexpr int BM = 128, BN = 128, BK = 32;
static constexpr int LDSROW = BK + 8;          // 64B data + 16B pad per row

union Frag {
    v16bf v;
    uint4 q[2];
};

// C[b, e, n] = act( A[b, slot(e), :] @ W[e] + bias[e] )
// A row-major (strideA elems/row), W[e] is K x ldw row-major, C strideC elems/row.
template <bool A_F32, bool OUT_BF16, bool RELU>
__global__ __launch_bounds__(256) void moe_gemm(
    const void* __restrict__ Av, long strideA, int slotShift, int slotMulK,
    const float* __restrict__ W, long wExpStride, int ldw,
    const float* __restrict__ bias, int biasExpStride,
    void* __restrict__ Cv, long strideC, int cExpStride, int K)
{
    __shared__ __bf16 sA[BM * LDSROW];
    __shared__ __bf16 sB[BN * LDSROW];         // transposed: [n][k]

    const int tid  = threadIdx.x;
    const int e    = blockIdx.z;
    const int n0   = blockIdx.x * BN;
    const int b0   = blockIdx.y * BM;
    const int lane = tid & 31;
    const int wave = tid >> 5;
    const int half = lane >> 4;
    const int lm   = lane & 15;

    const int  slot = e >> slotShift;
    const long aOff = (long)slot * slotMulK;

    const float*          Af = (const float*)Av;
    const unsigned short* Ab = (const unsigned short*)Av;

#if HAVE_TDM
    constexpr bool USE_TDM = !A_F32;           // bf16 A: pure DMA, no conversion
#else
    constexpr bool USE_TDM = false;
#endif

    v8f acc[8] = {};

    for (int k0 = 0; k0 < K; k0 += BK) {
        // ---- stage A tile (128 x 32) as bf16 ----
        if constexpr (USE_TDM) {
#if HAVE_TDM
            if (wave == 0) {
                // Tensor DMA descriptor (ISA 8.3/8.4): 2D tile 32 x 128 of 2B
                // elements, row stride = strideA, LDS pad 4 dwords every 16.
                const unsigned long long gaddr =
                    (unsigned long long)(uintptr_t)(Ab + (long)b0 * strideA + aOff + k0);
                const unsigned ldsOff = (unsigned)(uintptr_t)(&sA[0]);
                u32x4 g0;
                g0[0] = 1u;                                        // count=1
                g0[1] = ldsOff;                                    // lds_addr
                g0[2] = (unsigned)(gaddr & 0xFFFFFFFFull);         // global_addr lo
                g0[3] = (unsigned)((gaddr >> 32) & 0x01FFFFFFull)  // global_addr hi
                        | (2u << 30);                              // type = 2 (image)
                i32x8 g1;
                g1[0] = (1 << 16)       // data_size = 2 bytes
                      | (1 << 20)       // pad_enable
                      | (3 << 22)       // pad_interval: 16 dwords (64B)
                      | (3 << 25);      // pad_amount:   4 dwords (16B)
                g1[1] = (int)(32u << 16);                    // tensor_dim0 = 32 (lo16)
                g1[2] = (int)(128u << 16);                   // dim0 hi=0 | tensor_dim1=128 lo
                g1[3] = (int)(32u << 16);                    // dim1 hi=0 | tile_dim0 = 32
                g1[4] = (int)(128u);                         // tile_dim1 = 128, tile_dim2 = 0
                g1[5] = (int)((unsigned long long)strideA & 0xFFFFFFFFull); // dim0_stride lo
                g1[6] = (int)(((unsigned long long)strideA >> 32) & 0xFFFFull); // hi | dim1_stride lo = 0
                g1[7] = 0;
                i32x4 z4 = {0, 0, 0, 0};
                i32x8 z8 = {0, 0, 0, 0, 0, 0, 0, 0};
                __builtin_amdgcn_tensor_load_to_lds(g0, g1, z4, z4, z8, 0);
            }
#endif
        } else {
            const int  r    = tid >> 1;
            const int  part = tid & 1;
            const long g0i  = (long)(b0 + r) * strideA + aOff + k0 + part * 16;
            uint4 q0, q1;
            if constexpr (A_F32) {
                const float4* p = (const float4*)(Af + g0i);
                float4 f0 = p[0], f1 = p[1], f2 = p[2], f3 = p[3];
                unsigned pk[8];
                pk[0] = f2bf(f0.x) | ((unsigned)f2bf(f0.y) << 16);
                pk[1] = f2bf(f0.z) | ((unsigned)f2bf(f0.w) << 16);
                pk[2] = f2bf(f1.x) | ((unsigned)f2bf(f1.y) << 16);
                pk[3] = f2bf(f1.z) | ((unsigned)f2bf(f1.w) << 16);
                pk[4] = f2bf(f2.x) | ((unsigned)f2bf(f2.y) << 16);
                pk[5] = f2bf(f2.z) | ((unsigned)f2bf(f2.w) << 16);
                pk[6] = f2bf(f3.x) | ((unsigned)f2bf(f3.y) << 16);
                pk[7] = f2bf(f3.z) | ((unsigned)f2bf(f3.w) << 16);
                q0 = make_uint4(pk[0], pk[1], pk[2], pk[3]);
                q1 = make_uint4(pk[4], pk[5], pk[6], pk[7]);
            } else {
                const uint4* p = (const uint4*)(Ab + g0i);
                q0 = p[0];
                q1 = p[1];
            }
            uint4* dst = (uint4*)&sA[r * LDSROW + part * 16];
            dst[0] = q0;
            dst[1] = q1;
        }

        // ---- stage B tile (32 x 128) transposed -> sB[n][k], bf16 ----
        {
            const int n     = tid & 127;
            const int seg   = tid >> 7;        // 0..1
            const int kbase = seg * 16;
            const float* wp = W + wExpStride * e + (long)(k0 + kbase) * ldw + n0 + n;
            unsigned pk[8];
#pragma unroll
            for (int qq = 0; qq < 8; ++qq) {
                float f0 = wp[(2 * qq + 0) * ldw];
                float f1 = wp[(2 * qq + 1) * ldw];
                pk[qq] = f2bf(f0) | ((unsigned)f2bf(f1) << 16);
            }
            uint4* dst = (uint4*)&sB[n * LDSROW + kbase];
            dst[0] = make_uint4(pk[0], pk[1], pk[2], pk[3]);
            dst[1] = make_uint4(pk[4], pk[5], pk[6], pk[7]);
        }

#if HAVE_TDM
        if constexpr (USE_TDM) {
            if (wave == 0) __builtin_amdgcn_s_wait_tensorcnt(0);
        }
#endif
        __syncthreads();

        // ---- WMMA: wave owns rows [16*wave, 16*wave+16), all 128 cols ----
        Frag af;
        {
            const __bf16* pa = &sA[(16 * wave + lm) * LDSROW];
            af.q[0] = *(const uint4*)(pa + 8 * half);        // K 0-7  / 8-15
            af.q[1] = *(const uint4*)(pa + 16 + 8 * half);   // K 16-23/ 24-31
        }
#pragma unroll
        for (int j = 0; j < 8; ++j) {
            Frag bf_;
            const __bf16* pb = &sB[(16 * j + lm) * LDSROW];
            bf_.q[0] = *(const uint4*)(pb + 8 * half);
            bf_.q[1] = *(const uint4*)(pb + 16 + 8 * half);
            acc[j] = __builtin_amdgcn_wmma_f32_16x16x32_bf16(
                false, af.v, false, bf_.v, (short)0, acc[j], false, false);
        }
        __syncthreads();   // also WAR-protects next iteration's TDM write
    }

    // ---- epilogue: bias (+relu), store ----
    const int mbase = b0 + 16 * wave;
#pragma unroll
    for (int j = 0; j < 8; ++j) {
        const int   n  = n0 + 16 * j + lm;
        const float bv = bias[(long)biasExpStride * e + n];
#pragma unroll
        for (int v = 0; v < 8; ++v) {
            const int r   = mbase + v + 8 * half;
            float     val = acc[j][v] + bv;
            if (RELU) val = fmaxf(val, 0.0f);
            const long ci = (long)r * strideC + (long)e * cExpStride + n;
            if (OUT_BF16) ((unsigned short*)Cv)[ci] = f2bf(val);
            else          ((float*)Cv)[ci] = val;
        }
    }
}

// g[b,m,:] = softmax(x[b,m,:] @ Wg[m] + bg[m]) + structured add.  One wave per row.
__global__ __launch_bounds__(256) void gate_kernel(
    const float* __restrict__ x, const float* __restrict__ Wg,
    const float* __restrict__ bg, const float* __restrict__ sew_task,
    const float* __restrict__ sew_shared, float* __restrict__ g, int M, int layer)
{
    const int wave = threadIdx.x >> 5;
    const int lane = threadIdx.x & 31;
    int  id    = blockIdx.x * 8 + wave;
    bool valid = id < 1024 * M;
    if (!valid) id = 1024 * M - 1;
    const int b = id / M;
    const int m = id % M;

    const float* xr = x + ((long)b * 9 + m) * 512;
    const int    wl = lane < 18 ? lane : 17;
    const float* wg = Wg + (long)m * 512 * 18 + wl;

    float acc = 0.0f;
    for (int d = 0; d < 512; ++d)
        acc = fmaf(xr[d], wg[(long)d * 18], acc);

    float logit = (lane < 18) ? (acc + bg[m * 18 + lane]) : -3.0e38f;
    float mx = logit;
#pragma unroll
    for (int off = 16; off; off >>= 1) mx = fmaxf(mx, __shfl_xor(mx, off, 32));
    float ex = (lane < 18) ? __expf(logit - mx) : 0.0f;
    float s  = ex;
#pragma unroll
    for (int off = 16; off; off >>= 1) s += __shfl_xor(s, off, 32);
    float gv = ex / s;

    const int c0 = 2 * m;
    if (lane == c0 || lane == c0 + 1) {
        float a = (m < 8) ? sew_task[(m * 2 + layer) * 2 + (lane - c0)]
                          : sew_shared[lane - c0];
        gv += a;
    }
    if (valid && lane < 18) g[(long)id * 18 + lane] = gv;
}

// out[b,m,d] = sum_e g[b,m,e] * eo[b,e,d]
__global__ __launch_bounds__(256) void combine_kernel(
    const float* __restrict__ g, const float* __restrict__ eo,
    float* __restrict__ out, int M)
{
    const long idx   = (long)blockIdx.x * 256 + threadIdx.x;
    const long total = (long)1024 * M * 512;
    if (idx >= total) return;
    const int  d  = (int)(idx & 511);
    const int  m  = (int)((idx >> 9) % M);
    const long b  = idx / ((long)512 * M);
    const float* gr = g + (b * M + m) * 18;
    const float* er = eo + (b * 18) * 512 + d;
    float s = 0.0f;
#pragma unroll
    for (int e = 0; e < 18; ++e) s = fmaf(gr[e], er[(long)e * 512], s);
    out[idx] = s;
}

extern "C" void kernel_launch(void* const* d_in, const int* in_sizes, int n_in,
                              void* d_out, int out_size, void* d_ws, size_t ws_size,
                              hipStream_t stream)
{
    (void)in_sizes; (void)n_in; (void)out_size; (void)ws_size;
    const float* inputs     = (const float*)d_in[0];
    const float* W1_0       = (const float*)d_in[1];
    const float* b1_0       = (const float*)d_in[2];
    const float* W2_0       = (const float*)d_in[3];
    const float* b2_0       = (const float*)d_in[4];
    const float* Wg_0       = (const float*)d_in[5];
    const float* bg_0       = (const float*)d_in[6];
    const float* W1_1       = (const float*)d_in[7];
    const float* b1_1       = (const float*)d_in[8];
    const float* W2_1       = (const float*)d_in[9];
    const float* b2_1       = (const float*)d_in[10];
    const float* Wg_1       = (const float*)d_in[11];
    const float* bg_1       = (const float*)d_in[12];
    const float* sew_task   = (const float*)d_in[13];
    const float* sew_shared = (const float*)d_in[14];

    char* ws = (char*)d_ws;
    float*          gbuf = (float*)ws;                            //    663,552 B
    unsigned short* H1   = (unsigned short*)(ws + 663552);        // 37,748,736 B (bf16)
    float*          eo   = (float*)(ws + 663552 + 37748736);      // 37,748,736 B
    float*          x1   = (float*)(ws + 663552 + 2L * 37748736); // 18,874,368 B

    const dim3 blk(256);

    // ---------------- layer 0 (M = 9) ----------------
    gate_kernel<<<(1024 * 9 + 7) / 8, blk, 0, stream>>>(
        inputs, Wg_0, bg_0, sew_task, sew_shared, gbuf, 9, 0);

    // FFN1: h = relu(x @ W1 + b1)  -> H1 bf16   (K=512, N=1024)
    moe_gemm<true, true, true><<<dim3(1024 / BN, 1024 / BM, 18), blk, 0, stream>>>(
        inputs, 9 * 512, 1, 512,
        W1_0, (long)512 * 1024, 1024,
        b1_0, 1024,
        H1, 18 * 1024, 1024, 512);

    // FFN2: eo = h @ W2 + b2 -> eo f32          (K=1024, N=512), A via TDM
    moe_gemm<false, false, false><<<dim3(512 / BN, 1024 / BM, 18), blk, 0, stream>>>(
        H1, 18 * 1024, 0, 1024,
        W2_0, (long)1024 * 512, 512,
        b2_0, 512,
        eo, 18 * 512, 512, 1024);

    combine_kernel<<<(1024 * 9 * 512) / 256, blk, 0, stream>>>(gbuf, eo, x1, 9);

    // ---------------- layer 1 (M = 8) ----------------
    gate_kernel<<<(1024 * 8 + 7) / 8, blk, 0, stream>>>(
        x1, Wg_1, bg_1, sew_task, sew_shared, gbuf, 8, 1);

    moe_gemm<true, true, true><<<dim3(1024 / BN, 1024 / BM, 18), blk, 0, stream>>>(
        x1, 9 * 512, 1, 512,
        W1_1, (long)512 * 1024, 1024,
        b1_1, 1024,
        H1, 18 * 1024, 1024, 512);

    moe_gemm<false, false, false><<<dim3(512 / BN, 1024 / BM, 18), blk, 0, stream>>>(
        H1, 18 * 1024, 0, 1024,
        W2_1, (long)1024 * 512, 512,
        b2_1, 512,
        eo, 18 * 512, 512, 1024);

    combine_kernel<<<(1024 * 8 * 512) / 256, blk, 0, stream>>>(
        gbuf, eo, (float*)d_out, 8);
}
